// HeteroGNN_50199577755960
// MI455X (gfx1250) — compile-verified, matrix operand order checked by reading, not run on hardware
//
#include <hip/hip_runtime.h>
#include <hip/hip_bf16.h>

typedef __attribute__((ext_vector_type(16))) __bf16 v16bf;
typedef __attribute__((ext_vector_type(8)))  float  v8f;

#define NNODE 100000
#define NEDGE 2000000
#define NLBL  1000000
#define NEG_SLOPE 0.01f
#define BN_EPS 1e-5f

static __device__ __forceinline__ float atomAddF(float* p, float v) {
  return unsafeAtomicAdd(p, v);   // hardware global_atomic_add_f32
}

// ---------------------------------------------------------------- utilities
__global__ void zero_f32(float* __restrict__ p, long long n) {
  long long i = (long long)blockIdx.x * blockDim.x + threadIdx.x;
  if (i < n) p[i] = 0.0f;
}

__global__ void deg_count(const int* __restrict__ dst, float* __restrict__ deg, int E) {
  int e = blockIdx.x * blockDim.x + threadIdx.x;
  if (e < E) atomAddF(&deg[dst[e]], 1.0f);
}

__global__ void inv_deg(float* __restrict__ deg, int N) {
  int i = blockIdx.x * blockDim.x + threadIdx.x;
  if (i < N) deg[i] = 1.0f / fmaxf(deg[i], 1.0f);
}

// ------------------------------------------- weight folding:  Wc(K x 32), bc(32)
// h = x_dst @ (Wd@Wu_top) + aggmean @ (Ws@Wu_bot) + (bd@Wu_top + bs@Wu_bot + bu)
__global__ void prep_combine(const float* __restrict__ Ws, const float* __restrict__ bs,
                             const float* __restrict__ Wd, const float* __restrict__ bd,
                             const float* __restrict__ Wu, const float* __restrict__ bu,
                             float* __restrict__ Wc, float* __restrict__ bc, int F) {
  int tid = blockIdx.x * blockDim.x + threadIdx.x;
  int Ktot = 2 * F;
  if (tid < Ktot * 32) {
    int k = tid >> 5, n = tid & 31;
    float acc = 0.0f;
    if (k < F) {
      for (int j = 0; j < 32; ++j) acc += Wd[k * 32 + j] * Wu[j * 32 + n];
    } else {
      int kk = k - F;
      for (int j = 0; j < 32; ++j) acc += Ws[kk * 32 + j] * Wu[(32 + j) * 32 + n];
    }
    Wc[tid] = acc;
  }
  if (tid < 32) {
    float acc = bu[tid];
    for (int j = 0; j < 32; ++j)
      acc += bd[j] * Wu[j * 32 + tid] + bs[j] * Wu[(32 + j) * 32 + tid];
    bc[tid] = acc;
  }
}

// Pack Wc (K x 32 f32, row-major) into the CDNA5 16-bit B-fragment lane layout.
// Linear layout: Bp[((ktile*2 + half)*32 + lane)*16 + e]
__global__ void pack_b(const float* __restrict__ Wc, __bf16* __restrict__ Bp, int Ktot) {
  int tid = blockIdx.x * blockDim.x + threadIdx.x;
  if (tid >= Ktot * 32) return;
  int e    = tid & 15;
  int lane = (tid >> 4) & 31;
  int half = (tid >> 9) & 1;
  int kt   = tid >> 10;
  int n     = half * 16 + (lane & 15);
  int khalf = (lane >> 4) * 8;
  int j = e >> 1, t = e & 1;
  int k = kt * 32 + khalf + ((j < 4) ? 2 * j : 16 + 2 * (j - 4)) + t;
  Bp[tid] = (__bf16)Wc[k * 32 + n];
}

// ---------------------------------------------------------------- scatter-add
// thread = (edge, 4-feature group); float4 gather + 4 f32 atomics (L2-resident)
__global__ void scatter_add(const int* __restrict__ src, const int* __restrict__ dst,
                            const float* __restrict__ X, int xStride,
                            float* __restrict__ Agg, int aggStride,
                            int gShift, long long total) {
  long long tid = (long long)blockIdx.x * blockDim.x + threadIdx.x;
  if (tid >= total) return;
  long long e = tid >> gShift;
  int g = (int)(tid & ((1 << gShift) - 1));
  int s = src[e], d = dst[e];
  float4 x = *(const float4*)(X + (size_t)s * xStride + g * 4);
  float* a = Agg + (size_t)d * aggStride + g * 4;
  atomAddF(a + 0, x.x);
  atomAddF(a + 1, x.y);
  atomAddF(a + 2, x.z);
  atomAddF(a + 3, x.w);
}

// ------------------------------------- fused GEMM: Hpre = [Xdst | Agg/deg] @ B + bc
// One wave per 16x32 output tile; WMMA bf16 (K=32/step); BN stats via atomics.
// F is compile-time so the K loop fully unrolls and the x-vs-agg source select
// resolves statically (no EXEC-mask diamonds in the inner loop).
template <int F>
__global__ __launch_bounds__(256)
void gemm_bn_stats(const float* __restrict__ X,
                   const float* __restrict__ Agg,
                   const float* __restrict__ invdeg,
                   const __bf16* __restrict__ Bp,
                   const float* __restrict__ bc,
                   float* __restrict__ Hpre,
                   float* __restrict__ stats,
                   int nTiles) {
  constexpr int Ktot = 2 * F;
  int wave = threadIdx.x >> 5;
  int lane = threadIdx.x & 31;
  int tile = blockIdx.x * 8 + wave;
  if (tile >= nTiles) return;                 // wave-uniform exit (EXEC stays all-1 for WMMA)
  int rowBase = tile << 4;
  int mrow  = rowBase + (lane & 15);
  int khalf = (lane >> 4) << 3;               // 0 or 8 (16-bit A-fragment lane layout)
  float invd = invdeg[mrow];
  const float* xrow = X + (size_t)mrow * F + khalf;
  const float* grow = Agg + (size_t)mrow * F + khalf;

  v8f c0 = {};
  v8f c1 = {};
#pragma unroll
  for (int kt = 0; kt < Ktot; kt += 32) {
    float af[16];
#pragma unroll
    for (int seg = 0; seg < 2; ++seg) {
      const int k0 = kt + 16 * seg;           // compile-time constant
      const float* p = (k0 < F) ? (xrow + k0) : (grow + (k0 - F));
      const float s = (k0 < F) ? 1.0f : invd;
      float4 u = *(const float4*)p;
      float4 w = *(const float4*)(p + 4);
      af[8 * seg + 0] = u.x * s; af[8 * seg + 1] = u.y * s;
      af[8 * seg + 2] = u.z * s; af[8 * seg + 3] = u.w * s;
      af[8 * seg + 4] = w.x * s; af[8 * seg + 5] = w.y * s;
      af[8 * seg + 6] = w.z * s; af[8 * seg + 7] = w.w * s;
    }
    v16bf a;
#pragma unroll
    for (int i = 0; i < 16; ++i) a[i] = (__bf16)af[i];
    const v16bf* bp = ((const v16bf*)Bp) + (size_t)(kt >> 5) * 64 + lane;
    v16bf blo = bp[0];
    v16bf bhi = bp[32];
    c0 = __builtin_amdgcn_wmma_f32_16x16x32_bf16(false, a, false, blo, (short)0, c0, false, false);
    c1 = __builtin_amdgcn_wmma_f32_16x16x32_bf16(false, a, false, bhi, (short)0, c1, false, false);
  }

  int col0 = lane & 15;
  int col1 = col0 + 16;
  int rOff = (lane >> 4) << 3;                // C layout: lanes>=16 hold M=v+8
  float bb0 = bc[col0], bb1 = bc[col1];
  float s0 = 0, q0 = 0, s1 = 0, q1 = 0;
#pragma unroll
  for (int v = 0; v < 8; ++v) {
    int r = rowBase + rOff + v;
    float x0 = c0[v] + bb0;
    float x1 = c1[v] + bb1;
    Hpre[(size_t)r * 32 + col0] = x0;
    Hpre[(size_t)r * 32 + col1] = x1;
    s0 += x0; q0 += x0 * x0;
    s1 += x1; q1 += x1 * x1;
  }
  atomAddF(&stats[col0],      s0);
  atomAddF(&stats[col0 + 32], q0);
  atomAddF(&stats[col1],      s1);
  atomAddF(&stats[col1 + 32], q1);
}

// ---------------------------------------------------------------- BN + leaky
__global__ void bn_leaky(const float* __restrict__ Hpre, const float* __restrict__ stats,
                         const float* __restrict__ gma, const float* __restrict__ bta,
                         float* __restrict__ out, int N) {
  long long tid = (long long)blockIdx.x * blockDim.x + threadIdx.x;
  if (tid >= (long long)N * 32) return;
  int c = (int)(tid & 31);
  float invN = 1.0f / (float)N;
  float mu  = stats[c] * invN;
  float var = stats[32 + c] * invN - mu * mu;
  float xv = Hpre[tid];
  float y = gma[c] * (xv - mu) * rsqrtf(var + BN_EPS) + bta[c];
  out[tid] = (y >= 0.0f) ? y : NEG_SLOPE * y;
}

// ---------------------------------------------------------------- edge scoring
__global__ void edge_score(const int* __restrict__ lbl,
                           const float* __restrict__ HA, const float* __restrict__ HB,
                           float* __restrict__ out, int EL) {
  int e = blockIdx.x * blockDim.x + threadIdx.x;
  if (e >= EL) return;
  int ia = lbl[e];
  int ib = lbl[EL + e];
  const float4* pa = (const float4*)(HA + (size_t)ia * 32);
  const float4* pb = (const float4*)(HB + (size_t)ib * 32);
  float acc = 0.0f;
#pragma unroll
  for (int j = 0; j < 8; ++j) {
    float4 a = pa[j], b = pb[j];
    acc += a.x * b.x + a.y * b.y + a.z * b.z + a.w * b.w;
  }
  out[e] = acc;
}

// =====================================================================
extern "C" void kernel_launch(void* const* d_in, const int* in_sizes, int n_in,
                              void* d_out, int out_size, void* d_ws, size_t ws_size,
                              hipStream_t stream) {
  const int N = NNODE, E = NEDGE, EL = NLBL, F1 = 64, F2 = 32;

  const float* x_gene = (const float*)d_in[0];
  const float* x_dis  = (const float*)d_in[1];
  const int*   ei_g2d = (const int*)d_in[2];
  const int*   ei_d2g = (const int*)d_in[3];
  const int*   l_g2d  = (const int*)d_in[4];
  const int*   l_d2g  = (const int*)d_in[5];
  // params in insertion order; each conv = {Ws,bs,Wd,bd,Wu,bu}
  const float* const* P = (const float* const*)(d_in + 6);
  // l1.g2d: P[0..5]  l1.d2g: P[6..11]  l2.g2d: P[12..17]  l2.d2g: P[18..23]
  // bn1.gene g/b: P[24],P[25]  bn1.dis: P[26],P[27]  bn2.gene: P[28],P[29]  bn2.dis: P[30],P[31]
  float* out = (float*)d_out;

  // ---- workspace carve (64B aligned)
  char* wsb = (char*)d_ws;
  size_t off = 0;
  auto carve = [&](size_t bytes) -> char* {
    off = (off + 63) & ~(size_t)63;
    char* p = wsb + off;
    off += bytes;
    return p;
  };
  float* deg_d  = (float*)carve((size_t)N * 4);
  float* deg_g  = (float*)carve((size_t)N * 4);
  float* agg_d  = (float*)carve((size_t)N * 64 * 4);
  float* agg_g  = (float*)carve((size_t)N * 64 * 4);
  float* hpre_g = (float*)carve((size_t)N * 32 * 4);
  float* hpre_d = (float*)carve((size_t)N * 32 * 4);
  float* hg     = (float*)carve((size_t)N * 32 * 4);
  float* hd     = (float*)carve((size_t)N * 32 * 4);
  float* Wc1_gd = (float*)carve(128 * 32 * 4);
  float* Wc1_dg = (float*)carve(128 * 32 * 4);
  float* Wc2_gd = (float*)carve(64 * 32 * 4);
  float* Wc2_dg = (float*)carve(64 * 32 * 4);
  float* bc1_gd = (float*)carve(32 * 4);
  float* bc1_dg = (float*)carve(32 * 4);
  float* bc2_gd = (float*)carve(32 * 4);
  float* bc2_dg = (float*)carve(32 * 4);
  float* stats_g = (float*)carve(64 * 4);
  float* stats_d = (float*)carve(64 * 4);
  __bf16* Bp1_gd = (__bf16*)carve(128 * 32 * 2);
  __bf16* Bp1_dg = (__bf16*)carve(128 * 32 * 2);
  __bf16* Bp2_gd = (__bf16*)carve(64 * 32 * 2);
  __bf16* Bp2_dg = (__bf16*)carve(64 * 32 * 2);

  const int TB = 256;
  auto blocks = [&](long long n) -> unsigned { return (unsigned)((n + TB - 1) / TB); };

  // ---- degrees (shared by both layers; edge lists identical)
  zero_f32<<<blocks(N), TB, 0, stream>>>(deg_d, N);
  zero_f32<<<blocks(N), TB, 0, stream>>>(deg_g, N);
  deg_count<<<blocks(E), TB, 0, stream>>>(ei_g2d + E, deg_d, E);
  deg_count<<<blocks(E), TB, 0, stream>>>(ei_d2g + E, deg_g, E);
  inv_deg<<<blocks(N), TB, 0, stream>>>(deg_d, N);
  inv_deg<<<blocks(N), TB, 0, stream>>>(deg_g, N);

  // ---- fold weights, pack B fragments
  prep_combine<<<blocks(128 * 32), TB, 0, stream>>>(P[0], P[1], P[2], P[3], P[4], P[5],   Wc1_gd, bc1_gd, F1);
  prep_combine<<<blocks(128 * 32), TB, 0, stream>>>(P[6], P[7], P[8], P[9], P[10], P[11], Wc1_dg, bc1_dg, F1);
  prep_combine<<<blocks(64 * 32), TB, 0, stream>>>(P[12], P[13], P[14], P[15], P[16], P[17], Wc2_gd, bc2_gd, F2);
  prep_combine<<<blocks(64 * 32), TB, 0, stream>>>(P[18], P[19], P[20], P[21], P[22], P[23], Wc2_dg, bc2_dg, F2);
  pack_b<<<blocks(128 * 32), TB, 0, stream>>>(Wc1_gd, Bp1_gd, 128);
  pack_b<<<blocks(128 * 32), TB, 0, stream>>>(Wc1_dg, Bp1_dg, 128);
  pack_b<<<blocks(64 * 32), TB, 0, stream>>>(Wc2_gd, Bp2_gd, 64);
  pack_b<<<blocks(64 * 32), TB, 0, stream>>>(Wc2_dg, Bp2_dg, 64);

  const int nTiles = N / 16;                 // 6250 exactly
  const unsigned gemmGrid = (unsigned)((nTiles + 7) / 8);

  // ======== layer 1 ========
  zero_f32<<<blocks((long long)N * 64), TB, 0, stream>>>(agg_d, (long long)N * 64);
  zero_f32<<<blocks((long long)N * 64), TB, 0, stream>>>(agg_g, (long long)N * 64);
  scatter_add<<<blocks((long long)E * 16), TB, 0, stream>>>(ei_g2d, ei_g2d + E, x_gene, 64, agg_d, 64, 4, (long long)E * 16);
  scatter_add<<<blocks((long long)E * 16), TB, 0, stream>>>(ei_d2g, ei_d2g + E, x_dis, 64, agg_g, 64, 4, (long long)E * 16);
  zero_f32<<<1, TB, 0, stream>>>(stats_d, 64);
  zero_f32<<<1, TB, 0, stream>>>(stats_g, 64);
  gemm_bn_stats<64><<<gemmGrid, TB, 0, stream>>>(x_dis,  agg_d, deg_d, Bp1_gd, bc1_gd, hpre_d, stats_d, nTiles);
  gemm_bn_stats<64><<<gemmGrid, TB, 0, stream>>>(x_gene, agg_g, deg_g, Bp1_dg, bc1_dg, hpre_g, stats_g, nTiles);
  bn_leaky<<<blocks((long long)N * 32), TB, 0, stream>>>(hpre_g, stats_g, P[24], P[25], hg, N);
  bn_leaky<<<blocks((long long)N * 32), TB, 0, stream>>>(hpre_d, stats_d, P[26], P[27], hd, N);

  // ======== layer 2 ========
  zero_f32<<<blocks((long long)N * 32), TB, 0, stream>>>(agg_d, (long long)N * 32);
  zero_f32<<<blocks((long long)N * 32), TB, 0, stream>>>(agg_g, (long long)N * 32);
  scatter_add<<<blocks((long long)E * 8), TB, 0, stream>>>(ei_g2d, ei_g2d + E, hg, 32, agg_d, 32, 3, (long long)E * 8);
  scatter_add<<<blocks((long long)E * 8), TB, 0, stream>>>(ei_d2g, ei_d2g + E, hd, 32, agg_g, 32, 3, (long long)E * 8);
  zero_f32<<<1, TB, 0, stream>>>(stats_d, 64);
  zero_f32<<<1, TB, 0, stream>>>(stats_g, 64);
  gemm_bn_stats<32><<<gemmGrid, TB, 0, stream>>>(hd, agg_d, deg_d, Bp2_gd, bc2_gd, hpre_d, stats_d, nTiles);
  gemm_bn_stats<32><<<gemmGrid, TB, 0, stream>>>(hg, agg_g, deg_g, Bp2_dg, bc2_dg, hpre_g, stats_g, nTiles);
  bn_leaky<<<blocks((long long)N * 32), TB, 0, stream>>>(hpre_g, stats_g, P[28], P[29], hg, N);
  bn_leaky<<<blocks((long long)N * 32), TB, 0, stream>>>(hpre_d, stats_d, P[30], P[31], hd, N);

  // ======== edge scoring ========
  edge_score<<<blocks(EL), TB, 0, stream>>>(l_g2d, hg, hd, out, EL);
  edge_score<<<blocks(EL), TB, 0, stream>>>(l_d2g, hd, hg, out + EL, EL);
}
